// Hierarchical_Granular_Structural_Attention_31198642438650
// MI455X (gfx1250) — matrix operand order repaired and checked by reading, not compiled, not run on hardware
//
#include <hip/hip_runtime.h>
#include <cstdint>
#include <cstddef>

// ---------------------------------------------------------------------------
// Problem constants (match reference)
// ---------------------------------------------------------------------------
constexpr int Bc = 16384;   // batch rows
constexpr int Dc = 4096;    // features
constexpr int Hc = 32;      // heads
constexpr int HDc = 128;    // head dim
constexpr float EPSV = 1e-5f;

typedef __attribute__((ext_vector_type(16))) __bf16 v16bf;
typedef __attribute__((ext_vector_type(8)))  __bf16 v8bf;
typedef __attribute__((ext_vector_type(4)))  __bf16 bf4;
typedef __attribute__((ext_vector_type(8)))  float  v8f;

union AFrag { v16bf v; v8bf h[2]; };

// ---------------------------------------------------------------------------
// fp32 -> bf16 elementwise cast (weights)
// ---------------------------------------------------------------------------
__global__ __launch_bounds__(256)
void cast_f32_to_bf16(const float* __restrict__ in, __bf16* __restrict__ out, size_t n) {
  size_t i = ((size_t)blockIdx.x * 256 + threadIdx.x) * 4;
  if (i >= n) return;
  float4 v = *(const float4*)(in + i);
  bf4 o;
  o.x = (__bf16)v.x; o.y = (__bf16)v.y; o.z = (__bf16)v.z; o.w = (__bf16)v.w;
  *(bf4*)(out + i) = o;
}

// ---------------------------------------------------------------------------
// Row mean over features: xm[b] = mean(x[b,:])
// ---------------------------------------------------------------------------
__global__ __launch_bounds__(256)
void row_mean_kernel(const float* __restrict__ x, float* __restrict__ xm) {
  const int b = blockIdx.x;
  const int t = threadIdx.x;
  const float4* x4 = (const float4*)(x + (size_t)b * Dc);
  float s = 0.f;
  for (int j = t; j < Dc / 4; j += 256) {
    float4 v = x4[j];
    s += v.x + v.y + v.z + v.w;
  }
#pragma unroll
  for (int m = 16; m >= 1; m >>= 1) s += __shfl_xor(s, m, 32);
  __shared__ float red[8];
  if ((t & 31) == 0) red[t >> 5] = s;
  __syncthreads();
  if (t == 0) {
    float a = 0.f;
#pragma unroll
    for (int w = 0; w < 8; ++w) a += red[w];
    xm[b] = a / (float)Dc;
  }
}

// ---------------------------------------------------------------------------
// Column sums over batch (partial, atomic accumulate): ymsum[d] += sum_r x[r,d]
// ---------------------------------------------------------------------------
__global__ __launch_bounds__(256)
void col_sum_kernel(const float* __restrict__ x, float* __restrict__ ymsum) {
  const int d = blockIdx.x * 256 + threadIdx.x;
  const int r0 = blockIdx.y * 256;
  float s = 0.f;
  for (int r = 0; r < 256; ++r) s += x[(size_t)(r0 + r) * Dc + d];
  atomicAdd(&ymsum[d], s);
}

// ---------------------------------------------------------------------------
// Vector mean/var (single block): out2 = {mean(u), var(u)}, u = v * scale
// ---------------------------------------------------------------------------
__global__ __launch_bounds__(256)
void vec_stats_kernel(const float* __restrict__ v, int n, float scale,
                      float* __restrict__ out2) {
  const int t = threadIdx.x;
  float s1 = 0.f, s2 = 0.f;
  for (int i = t; i < n; i += 256) {
    float u = v[i] * scale;
    s1 += u; s2 += u * u;
  }
#pragma unroll
  for (int m = 16; m >= 1; m >>= 1) {
    s1 += __shfl_xor(s1, m, 32);
    s2 += __shfl_xor(s2, m, 32);
  }
  __shared__ float r1[8], r2[8];
  if ((t & 31) == 0) { r1[t >> 5] = s1; r2[t >> 5] = s2; }
  __syncthreads();
  if (t == 0) {
    float a = 0.f, c = 0.f;
#pragma unroll
    for (int w = 0; w < 8; ++w) { a += r1[w]; c += r2[w]; }
    float mean = a / (float)n;
    out2[0] = mean;
    out2[1] = c / (float)n - mean * mean;
  }
}

// ---------------------------------------------------------------------------
// Gate: gout[i] = sigmoid( myln(w*u + b) ), stats of (w*u+b) derived affinely
// myln: (c - mean_c) / (sqrt(var_c + eps) + eps)
// ---------------------------------------------------------------------------
__global__ __launch_bounds__(256)
void gate_kernel(const float* __restrict__ v, int n, float scale,
                 const float* __restrict__ wp, const float* __restrict__ bp,
                 const float* __restrict__ st, float* __restrict__ gout) {
  int i = blockIdx.x * 256 + threadIdx.x;
  if (i >= n) return;
  float w = wp[0], bb = bp[0];
  float u = v[i] * scale;
  float c = w * u + bb;
  float mc = w * st[0] + bb;
  float vc = w * w * st[1];
  float tn = (c - mc) / (sqrtf(vc + EPSV) + EPSV);
  gout[i] = 1.f / (1.f + __expf(-tn));
}

// ---------------------------------------------------------------------------
// g = x * gx[b] * gy[d], cast to bf16 (GEMM input)
// grid: (Bc, Dc/1024), block 256, 4 elems/thread
// ---------------------------------------------------------------------------
__global__ __launch_bounds__(256)
void build_g_bf16(const float* __restrict__ x, const float* __restrict__ gx,
                  const float* __restrict__ gy, __bf16* __restrict__ g) {
  const int b = blockIdx.x;
  const int c0 = (blockIdx.y * 256 + threadIdx.x) * 4;
  const float gxb = gx[b];
  float4 xv = *(const float4*)(x + (size_t)b * Dc + c0);
  float4 gv = *(const float4*)(gy + c0);
  bf4 o;
  o.x = (__bf16)(xv.x * gxb * gv.x);
  o.y = (__bf16)(xv.y * gxb * gv.y);
  o.z = (__bf16)(xv.z * gxb * gv.z);
  o.w = (__bf16)(xv.w * gxb * gv.w);
  *(bf4*)(g + (size_t)b * Dc + c0) = o;
}

// ---------------------------------------------------------------------------
// bf16 WMMA GEMM: C[M,N] = A[M,K] @ W[N,K]^T + bias[N]   (torch Linear)
// A, W bf16 row-major (both contiguous along K), C fp32 row-major.
// Block = 256 threads (8 waves), block tile 128(M) x 256(N), K-step 32.
// Wave grid 2(M) x 4(N): each wave -> 64x64 = 4x4 WMMA tiles (16 wmma/K-step).
// LDS double-buffered (2 x 24 KB), filled with CDNA5 async-to-LDS loads.
// ---------------------------------------------------------------------------
__global__ __launch_bounds__(256)
void wmma_gemm_bf16(const __bf16* __restrict__ A, const __bf16* __restrict__ W,
                    const float* __restrict__ bias, float* __restrict__ C,
                    int M, int N, int K) {
  constexpr int BK = 32;
  __shared__ __align__(64) __bf16 sA[2][128 * BK];   // 2 x 8 KB
  __shared__ __align__(64) __bf16 sB[2][256 * BK];   // 2 x 16 KB

  const int tid  = threadIdx.x;
  const int lane = tid & 31;
  const int wave = tid >> 5;          // 0..7
  const int waveM = wave >> 2;        // 0..1 -> 64-row slab
  const int waveN = wave & 3;         // 0..3 -> 64-col slab
  const int l16  = lane & 15;
  const int hsel = lane >> 4;         // 0/1: K-half select per ISA layout
  const int kb8  = hsel * 8;          // unused; A frag also uses kb16 K halves {0..15}/{16..31}
  const int kb16 = hsel * 16;         // B frag K half {0..15}/{16..31}

  const int row0 = blockIdx.y * 128;
  const int col0 = blockIdx.x * 256;

  const uint32_t sA_base = (uint32_t)(uintptr_t)&sA[0][0];
  const uint32_t sB_base = (uint32_t)(uintptr_t)&sB[0][0];

  // Cooperative async fill of one K-step tile pair (A: 128x32, W: 256x32).
  // A: 512 16-byte chunks (2/thread); W: 1024 chunks (4/thread).
  auto issue_tile_loads = [&](int buf, int kt) {
    const int k0 = kt * BK;
    const uint32_t la0 = sA_base + (uint32_t)buf * 8192u;
    const uint32_t lb0 = sB_base + (uint32_t)buf * 16384u;
#pragma unroll
    for (int j = 0; j < 2; ++j) {
      const int c   = tid + j * 256;      // chunk id 0..511
      const int r   = c >> 2;             // tile row
      const int seg = c & 3;              // 8-elem segment within 32-K row
      const uint64_t ga = (uint64_t)(uintptr_t)(A + (size_t)(row0 + r) * K + (k0 + seg * 8));
      const uint32_t la = la0 + (uint32_t)c * 16u;
      asm volatile("global_load_async_to_lds_b128 %0, %1, off"
                   :: "v"(la), "v"(ga) : "memory");
    }
#pragma unroll
    for (int j = 0; j < 4; ++j) {
      const int c   = tid + j * 256;      // chunk id 0..1023
      const int r   = c >> 2;
      const int seg = c & 3;
      const uint64_t gw = (uint64_t)(uintptr_t)(W + (size_t)(col0 + r) * K + (k0 + seg * 8));
      const uint32_t lb = lb0 + (uint32_t)c * 16u;
      asm volatile("global_load_async_to_lds_b128 %0, %1, off"
                   :: "v"(lb), "v"(gw) : "memory");
    }
  };

  v8f acc[4][4];
#pragma unroll
  for (int mt = 0; mt < 4; ++mt)
#pragma unroll
    for (int nt = 0; nt < 4; ++nt)
      acc[mt][nt] = (v8f){0.f, 0.f, 0.f, 0.f, 0.f, 0.f, 0.f, 0.f};

  const int NKT = K / BK;               // 128 K-steps
  issue_tile_loads(0, 0);

  for (int kt = 0; kt < NKT; ++kt) {
    const int buf = kt & 1;
    asm volatile("s_wait_asynccnt 0" ::: "memory");
    __syncthreads();                     // tiles(kt) visible; compute(kt-1) done
    if (kt + 1 < NKT) issue_tile_loads(buf ^ 1, kt + 1);

    const __bf16* sa = &sA[buf][0];
    const __bf16* sb = &sB[buf][0];

    v16bf af[4];
#pragma unroll
    for (int mt = 0; mt < 4; ++mt) {
      const int r = waveM * 64 + mt * 16 + l16;
      AFrag u;
      u.h[0] = *(const v8bf*)(sa + r * BK + kb16);
      u.h[1] = *(const v8bf*)(sa + r * BK + kb16 + 8);
      af[mt] = u.v;
    }
#pragma unroll
    for (int nt = 0; nt < 4; ++nt) {
      const int nr = waveN * 64 + nt * 16 + l16;
      v16bf bfr = *(const v16bf*)(sb + nr * BK + kb16);
#pragma unroll
      for (int mt = 0; mt < 4; ++mt)
        acc[mt][nt] = __builtin_amdgcn_wmma_f32_16x16x32_bf16(
            false, af[mt], false, bfr, (short)0, acc[mt][nt], false, false);
    }
  }

  // Epilogue: C/D layout -> VGPR r: lanes0-15 (M=r, N=l16), lanes16-31 (M=8+r)
  const int rb = row0 + waveM * 64;
  const int nb = col0 + waveN * 64;
#pragma unroll
  for (int mt = 0; mt < 4; ++mt) {
#pragma unroll
    for (int nt = 0; nt < 4; ++nt) {
      const int col = nb + nt * 16 + l16;
      const float bvv = bias[col];
#pragma unroll
      for (int r = 0; r < 8; ++r) {
        const int row = rb + mt * 16 + hsel * 8 + r;
        C[(size_t)row * N + col] = acc[mt][nt][r] + bvv;
      }
    }
  }
}

// ---------------------------------------------------------------------------
// Fused: per-head scores -> softmax over heads -> attn = w*V -> +g -> LN -> bf16
// One block per row b. g recomputed from x,gx,gy (cheaper than storing fp32 g).
// ---------------------------------------------------------------------------
__global__ __launch_bounds__(256)
void attn_softmax_ln_kernel(const float* __restrict__ Q, const float* __restrict__ Km,
                            const float* __restrict__ V, const float* __restrict__ x,
                            const float* __restrict__ gx, const float* __restrict__ gy,
                            const float* __restrict__ gamma, const float* __restrict__ beta,
                            __bf16* __restrict__ hout) {
  const int b = blockIdx.x;
  const int t = threadIdx.x;                 // 0..255, 16 elems each
  const size_t row = (size_t)b * Dc;
  const int base = t * 16;
  const int hh = t >> 3;                     // head of this thread (8 thr/head)

  // per-head Q.K dot
  const float4* Q4 = (const float4*)(Q + row);
  const float4* K4 = (const float4*)(Km + row);
  float qk = 0.f;
#pragma unroll
  for (int j = 0; j < 4; ++j) {
    float4 qv = Q4[t * 4 + j];
    float4 kv = K4[t * 4 + j];
    qk += qv.x * kv.x + qv.y * kv.y + qv.z * kv.z + qv.w * kv.w;
  }
#pragma unroll
  for (int m = 1; m < 8; m <<= 1) qk += __shfl_xor(qk, m, 32);

  __shared__ float ssc[Hc];
  __shared__ float sw[Hc];
  if ((t & 7) == 0) ssc[hh] = qk * 0.08838834764831845f;  // 1/sqrt(128)
  __syncthreads();

  if (t < 32) {                              // softmax over 32 heads in wave 0
    float s = ssc[t];
    float mx = s;
#pragma unroll
    for (int m = 16; m >= 1; m >>= 1) mx = fmaxf(mx, __shfl_xor(mx, m, 32));
    float e = __expf(s - mx);
    float sum = e;
#pragma unroll
    for (int m = 16; m >= 1; m >>= 1) sum += __shfl_xor(sum, m, 32);
    sw[t] = e / sum;
  }
  __syncthreads();

  const float wh = sw[hh];
  const float gxb = gx[b];
  const float4* V4 = (const float4*)(V + row);
  const float4* X4 = (const float4*)(x + row);
  const float4* GY4 = (const float4*)gy;
  float vals[16];
  float s1 = 0.f, s2 = 0.f;
#pragma unroll
  for (int j = 0; j < 4; ++j) {
    float4 vv = V4[t * 4 + j];
    float4 xv = X4[t * 4 + j];
    float4 gv = GY4[t * 4 + j];
    float a0 = wh * vv.x + xv.x * gxb * gv.x;
    float a1 = wh * vv.y + xv.y * gxb * gv.y;
    float a2 = wh * vv.z + xv.z * gxb * gv.z;
    float a3 = wh * vv.w + xv.w * gxb * gv.w;
    vals[j * 4 + 0] = a0; vals[j * 4 + 1] = a1;
    vals[j * 4 + 2] = a2; vals[j * 4 + 3] = a3;
    s1 += a0 + a1 + a2 + a3;
    s2 += a0 * a0 + a1 * a1 + a2 * a2 + a3 * a3;
  }
#pragma unroll
  for (int m = 16; m >= 1; m >>= 1) {
    s1 += __shfl_xor(s1, m, 32);
    s2 += __shfl_xor(s2, m, 32);
  }
  __shared__ float r1[8], r2[8], mv[2];
  if ((t & 31) == 0) { r1[t >> 5] = s1; r2[t >> 5] = s2; }
  __syncthreads();
  if (t == 0) {
    float a = 0.f, c = 0.f;
#pragma unroll
    for (int w = 0; w < 8; ++w) { a += r1[w]; c += r2[w]; }
    float mean = a / (float)Dc;
    float var = c / (float)Dc - mean * mean;
    mv[0] = mean;
    mv[1] = rsqrtf(var + EPSV);
  }
  __syncthreads();
  const float mean = mv[0], rstd = mv[1];
#pragma unroll
  for (int j = 0; j < 4; ++j) {
    float4 gm = *(const float4*)(gamma + base + j * 4);
    float4 bt = *(const float4*)(beta + base + j * 4);
    bf4 o;
    o.x = (__bf16)((vals[j * 4 + 0] - mean) * rstd * gm.x + bt.x);
    o.y = (__bf16)((vals[j * 4 + 1] - mean) * rstd * gm.y + bt.y);
    o.z = (__bf16)((vals[j * 4 + 2] - mean) * rstd * gm.z + bt.z);
    o.w = (__bf16)((vals[j * 4 + 3] - mean) * rstd * gm.w + bt.w);
    *(bf4*)(hout + row + base + j * 4) = o;
  }
}

// ---------------------------------------------------------------------------
// Launch
// ---------------------------------------------------------------------------
extern "C" void kernel_launch(void* const* d_in, const int* in_sizes, int n_in,
                              void* d_out, int out_size, void* d_ws, size_t ws_size,
                              hipStream_t stream) {
  (void)in_sizes; (void)n_in; (void)out_size; (void)ws_size;
  const float* x     = (const float*)d_in[0];
  const float* wx    = (const float*)d_in[1];
  const float* bx    = (const float*)d_in[2];
  const float* wy    = (const float*)d_in[3];
  const float* by    = (const float*)d_in[4];
  const float* Wq    = (const float*)d_in[5];
  const float* bq    = (const float*)d_in[6];
  const float* Wk    = (const float*)d_in[7];
  const float* bk    = (const float*)d_in[8];
  const float* Wv    = (const float*)d_in[9];
  const float* bv    = (const float*)d_in[10];
  const float* Wo    = (const float*)d_in[11];
  const float* bo    = (const float*)d_in[12];
  const float* gamma = (const float*)d_in[13];
  const float* beta  = (const float*)d_in[14];

  char* ws = (char*)d_ws;
  size_t off = 0;
  auto carve = [&](size_t bytes) -> char* {
    char* p = ws + off;
    off += (bytes + 255) & ~(size_t)255;
    return p;
  };
  float*  xm    = (float*)carve((size_t)Bc * 4);
  float*  ymsum = (float*)carve((size_t)Dc * 4);
  float*  gxv   = (float*)carve((size_t)Bc * 4);
  float*  gyv   = (float*)carve((size_t)Dc * 4);
  float*  statx = (float*)carve(64);
  float*  staty = (float*)carve(64);
  __bf16* Wq_bf = (__bf16*)carve((size_t)Dc * Dc * 2);
  __bf16* Wk_bf = (__bf16*)carve((size_t)Dc * Dc * 2);
  __bf16* Wv_bf = (__bf16*)carve((size_t)Dc * Dc * 2);
  __bf16* Wo_bf = (__bf16*)carve((size_t)Dc * Dc * 2);
  __bf16* g_bf  = (__bf16*)carve((size_t)Bc * Dc * 2);
  __bf16* h_bf  = (__bf16*)carve((size_t)Bc * Dc * 2);
  float*  Qb    = (float*)carve((size_t)Bc * Dc * 4);
  float*  Kb    = (float*)carve((size_t)Bc * Dc * 4);
  float*  Vb    = (float*)carve((size_t)Bc * Dc * 4);

  // 1) weights -> bf16
  const size_t wn = (size_t)Dc * Dc;
  dim3 cgrid((unsigned)(wn / 1024));
  cast_f32_to_bf16<<<cgrid, 256, 0, stream>>>(Wq, Wq_bf, wn);
  cast_f32_to_bf16<<<cgrid, 256, 0, stream>>>(Wk, Wk_bf, wn);
  cast_f32_to_bf16<<<cgrid, 256, 0, stream>>>(Wv, Wv_bf, wn);
  cast_f32_to_bf16<<<cgrid, 256, 0, stream>>>(Wo, Wo_bf, wn);

  // 2) GRA statistics
  row_mean_kernel<<<Bc, 256, 0, stream>>>(x, xm);
  (void)hipMemsetAsync(ymsum, 0, (size_t)Dc * 4, stream);
  col_sum_kernel<<<dim3(Dc / 256, Bc / 256), 256, 0, stream>>>(x, ymsum);
  vec_stats_kernel<<<1, 256, 0, stream>>>(xm, Bc, 1.f, statx);
  vec_stats_kernel<<<1, 256, 0, stream>>>(ymsum, Dc, 1.f / (float)Bc, staty);
  gate_kernel<<<Bc / 256, 256, 0, stream>>>(xm, Bc, 1.f, wx, bx, statx, gxv);
  gate_kernel<<<Dc / 256, 256, 0, stream>>>(ymsum, Dc, 1.f / (float)Bc, wy, by, staty, gyv);

  // 3) g = x * gx * gy (bf16)
  build_g_bf16<<<dim3(Bc, Dc / 1024), 256, 0, stream>>>(x, gxv, gyv, g_bf);

  // 4) Q/K/V GEMMs (bf16 WMMA, f32 accumulate)
  dim3 ggrid(Dc / 256, Bc / 128);   // (16, 128)
  wmma_gemm_bf16<<<ggrid, 256, 0, stream>>>(g_bf, Wq_bf, bq, Qb, Bc, Dc, Dc);
  wmma_gemm_bf16<<<ggrid, 256, 0, stream>>>(g_bf, Wk_bf, bk, Kb, Bc, Dc, Dc);
  wmma_gemm_bf16<<<ggrid, 256, 0, stream>>>(g_bf, Wv_bf, bv, Vb, Bc, Dc, Dc);

  // 5) fused scores/softmax/attn/residual/LN -> h (bf16)
  attn_softmax_ln_kernel<<<Bc, 256, 0, stream>>>(Qb, Kb, Vb, x, gxv, gyv, gamma, beta, h_bf);

  // 6) output GEMM -> d_out (fp32)
  wmma_gemm_bf16<<<ggrid, 256, 0, stream>>>(h_bf, Wo_bf, bo, (float*)d_out, Bc, Dc, Dc);
}
